// MultiHeadAttention_65498251264098
// MI455X (gfx1250) — matrix operand order repaired and checked by reading, not compiled, not run on hardware
//
#include <hip/hip_runtime.h>
#include <hip/hip_bf16.h>

// ---------------- problem constants ----------------
#define B_  2
#define T_  2048
#define C_  1024
#define H_  16
#define HD_ 64

typedef unsigned short u16;
typedef unsigned int   u32;
typedef __attribute__((ext_vector_type(16))) __bf16 bf16x16;
typedef __attribute__((ext_vector_type(8)))  float  floatx8;
typedef __attribute__((ext_vector_type(4)))  u32    u32x4;

union Frag16 { bf16x16 v; u32x4 q[2]; };

// fp32 -> bf16 round-to-nearest-even
__device__ __forceinline__ u16 f2bf(float f) {
  u32 u = __float_as_uint(f);
  u32 r = u + 0x7fffu + ((u >> 16) & 1u);
  return (u16)(r >> 16);
}

__device__ __forceinline__ floatx8 zero8() {
  floatx8 z;
#pragma unroll
  for (int i = 0; i < 8; ++i) z[i] = 0.0f;
  return z;
}

// A-matrix 16x32 bf16 fragment: lane L holds row (L&15), K =
// 2*kb + 0..15 contiguous, kb = (L<16)?0:8.  Two 16B loads.
__device__ __forceinline__ bf16x16 load_a_frag(const u16* row, int kb) {
  Frag16 f;
  f.q[0] = *reinterpret_cast<const u32x4*>(row + 2 * kb);
  f.q[1] = *reinterpret_cast<const u32x4*>(row + 2 * kb + 8);
  return f.v;
}

// B-matrix 32x16 bf16 fragment: lane L holds column (L&15),
// K = (L>>4)*16 + 0..15 contiguous.  One 32B load (caller adds koff).
__device__ __forceinline__ bf16x16 load_b_frag(const u16* row, int koff) {
  Frag16 f;
  const u32x4* p = reinterpret_cast<const u32x4*>(row + koff);
  f.q[0] = p[0];
  f.q[1] = p[1];
  return f.v;
}

__device__ __forceinline__ floatx8 wmma_bf16(bf16x16 a, bf16x16 b, floatx8 c) {
  return __builtin_amdgcn_wmma_f32_16x16x32_bf16(
      /*neg_a=*/false, a, /*neg_b=*/false, b,
      /*c_mod=*/(short)0, c, /*reuse_a=*/false, /*reuse_b=*/false);
}

// ---------------- stage 0: fp32 -> bf16 ----------------
__global__ void f32_to_bf16_kernel(const float* __restrict__ src,
                                   u16* __restrict__ dst, int n) {
  int i = blockIdx.x * blockDim.x + threadIdx.x;
  if (i < n) dst[i] = f2bf(src[i]);
}

// ---------------- stage 1/3: projection GEMM  y = x @ W.T ----------------
// One wave computes a 16(time) x 64(one head / o-tile) block.
// mode 0: q  -> rope -> qbf (bf16)
// mode 1: k  -> rope -> kbf (bf16) + kout (f32, d_out region)
// mode 2: v  -> vout (f32, d_out region) + vT bf16 [B,H,HD,T]
// mode 3: out projection -> outp (f32, d_out region), A = attn bf16 [B,T,C]
__global__ __launch_bounds__(128) void gemm16x64_kernel(
    const u16* __restrict__ Ab, const u16* __restrict__ Wb, int mode,
    u16* __restrict__ qbf, u16* __restrict__ kbf, u16* __restrict__ vTb,
    float* __restrict__ kout, float* __restrict__ vout,
    float* __restrict__ outp) {
  const int wid  = blockIdx.x * 4 + (threadIdx.x >> 5);
  const int lane = threadIdx.x & 31;
  const int nqt  = T_ / 16;                 // 128 row tiles
  const int qt   = wid % nqt;
  const int h    = (wid / nqt) % H_;        // head (or o-tile for mode 3)
  const int b    = wid / (nqt * H_);
  const int t0   = qt * 16;
  const int m    = lane & 15;
  const int half = lane >> 4;
  const int kb   = half ? 8 : 0;
  const int koffB = half * 16;
  const int obase = h * 64;

  const u16* arow = Ab + (size_t)(b * T_ + t0 + m) * C_;

  floatx8 acc0 = zero8(), acc1 = zero8(), acc2 = zero8(), acc3 = zero8();
  for (int k0 = 0; k0 < C_; k0 += 32) {
    bf16x16 af = load_a_frag(arow + k0, kb);
    const u16* w0 = Wb + (size_t)(obase + 0 * 16 + m) * C_ + k0;
    const u16* w1 = Wb + (size_t)(obase + 1 * 16 + m) * C_ + k0;
    const u16* w2 = Wb + (size_t)(obase + 2 * 16 + m) * C_ + k0;
    const u16* w3 = Wb + (size_t)(obase + 3 * 16 + m) * C_ + k0;
    acc0 = wmma_bf16(af, load_b_frag(w0, koffB), acc0);
    acc1 = wmma_bf16(af, load_b_frag(w1, koffB), acc1);
    acc2 = wmma_bf16(af, load_b_frag(w2, koffB), acc2);
    acc3 = wmma_bf16(af, load_b_frag(w3, koffB), acc3);
  }

  const int n = lane & 15;  // output column within a 16-wide tile
  if (mode == 0 || mode == 1) {
    // RoPE: cols 0..31 = (acc0,acc1), cols 32..63 = (acc2,acc3).
    // Pair (acc0,acc2): i=n ; pair (acc1,acc3): i=16+n.
    const float kln = 0.28782313662425572f;  // ln(10000)/32
    float invf0 = __expf(-(float)n * kln);
    float invf1 = __expf(-(float)(16 + n) * kln);
#pragma unroll
    for (int r = 0; r < 8; ++r) {
      int t = t0 + r + 8 * half;
      float s0, c0, s1, c1;
      __sincosf((float)t * invf0, &s0, &c0);
      __sincosf((float)t * invf1, &s1, &c1);
      float a0 = acc0[r], a1 = acc1[r], a2 = acc2[r], a3 = acc3[r];
      float r0 = a0 * c0 - a2 * s0;
      float r2 = a0 * s0 + a2 * c0;
      float r1 = a1 * c1 - a3 * s1;
      float r3 = a1 * s1 + a3 * c1;
      size_t base = ((size_t)((b * H_ + h) * T_ + t)) * HD_;
      if (mode == 0) {
        qbf[base + 0 + n]  = f2bf(r0);
        qbf[base + 16 + n] = f2bf(r1);
        qbf[base + 32 + n] = f2bf(r2);
        qbf[base + 48 + n] = f2bf(r3);
      } else {
        kbf[base + 0 + n]  = f2bf(r0);
        kbf[base + 16 + n] = f2bf(r1);
        kbf[base + 32 + n] = f2bf(r2);
        kbf[base + 48 + n] = f2bf(r3);
        kout[base + 0 + n]  = r0;
        kout[base + 16 + n] = r1;
        kout[base + 32 + n] = r2;
        kout[base + 48 + n] = r3;
      }
    }
  } else if (mode == 2) {
#pragma unroll
    for (int r = 0; r < 8; ++r) {
      int t = t0 + r + 8 * half;
      size_t base = ((size_t)((b * H_ + h) * T_ + t)) * HD_;
#pragma unroll
      for (int nt = 0; nt < 4; ++nt) {
        float vv = (nt == 0) ? acc0[r] : (nt == 1) ? acc1[r]
                 : (nt == 2) ? acc2[r] : acc3[r];
        int d = nt * 16 + n;
        vout[base + d] = vv;
        vTb[((size_t)((b * H_ + h) * HD_ + d)) * T_ + t] = f2bf(vv);
      }
    }
  } else {
#pragma unroll
    for (int r = 0; r < 8; ++r) {
      int t = t0 + r + 8 * half;
      size_t base = (size_t)(b * T_ + t) * C_ + obase;
      outp[base + 0 + n]  = acc0[r];
      outp[base + 16 + n] = acc1[r];
      outp[base + 32 + n] = acc2[r];
      outp[base + 48 + n] = acc3[r];
    }
  }
}

// ---------------- stage 2: causal flash attention ----------------
// One wave per (b, h, 16-query tile); streams keys 32 at a time.
__global__ __launch_bounds__(128) void flash_attn_kernel(
    const u16* __restrict__ qbf, const u16* __restrict__ kbf,
    const u16* __restrict__ vTb, u16* __restrict__ attnb) {
  __shared__ __align__(16) u16 Pb[4][16 * 32];

  const int wv   = threadIdx.x >> 5;
  const int wid  = blockIdx.x * 4 + wv;
  const int lane = threadIdx.x & 31;
  const int nqt  = T_ / 16;
  const int qt   = wid % nqt;
  const int h    = (wid / nqt) % H_;
  const int b    = wid / (nqt * H_);
  const int t0   = qt * 16;
  const int m    = lane & 15;
  const int n    = m;
  const int half = lane >> 4;
  const int kb   = half ? 8 : 0;
  const int koffB = half * 16;
  const float scale = 0.125f;  // 1/sqrt(64)

  const size_t headBase = (size_t)(b * H_ + h) * T_;
  const u16* qrow = qbf + (headBase + t0 + m) * HD_;
  bf16x16 aQ0 = load_a_frag(qrow, kb);        // d = 0..31
  bf16x16 aQ1 = load_a_frag(qrow + 32, kb);   // d = 32..63

  floatx8 o0 = zero8(), o1 = zero8(), o2 = zero8(), o3 = zero8();
  float rm[8], rl[8];
#pragma unroll
  for (int r = 0; r < 8; ++r) { rm[r] = -1e30f; rl[r] = 0.0f; }

  u16* P = Pb[wv];
  const u16* vhead = vTb + (size_t)(b * H_ + h) * HD_ * T_;

  for (int j0 = 0; j0 < t0 + 16; j0 += 32) {
    float e0[8], e1[8];
#pragma unroll
    for (int s = 0; s < 2; ++s) {
      const u16* krow = kbf + (headBase + j0 + s * 16 + n) * HD_;
      bf16x16 bK0 = load_b_frag(krow, koffB);
      bf16x16 bK1 = load_b_frag(krow + 32, koffB);
      floatx8 sfr = zero8();
      sfr = wmma_bf16(aQ0, bK0, sfr);
      sfr = wmma_bf16(aQ1, bK1, sfr);
      int j = j0 + s * 16 + n;
#pragma unroll
      for (int r = 0; r < 8; ++r) {
        int t = t0 + r + 8 * half;
        float ev = sfr[r] * scale;
        ev = (j <= t) ? ev : -1e30f;
        if (s == 0) e0[r] = ev; else e1[r] = ev;
      }
    }
    // online softmax (rows live inside one 16-lane half)
#pragma unroll
    for (int r = 0; r < 8; ++r) {
      float mx = fmaxf(e0[r], e1[r]);
      mx = fmaxf(mx, __shfl_xor(mx, 1, 32));
      mx = fmaxf(mx, __shfl_xor(mx, 2, 32));
      mx = fmaxf(mx, __shfl_xor(mx, 4, 32));
      mx = fmaxf(mx, __shfl_xor(mx, 8, 32));
      float nm  = fmaxf(rm[r], mx);
      float sc2 = __expf(rm[r] - nm);
      rm[r] = nm;
      float p0 = __expf(e0[r] - nm);
      float p1 = __expf(e1[r] - nm);
      float ss = p0 + p1;
      ss += __shfl_xor(ss, 1, 32);
      ss += __shfl_xor(ss, 2, 32);
      ss += __shfl_xor(ss, 4, 32);
      ss += __shfl_xor(ss, 8, 32);
      rl[r] = rl[r] * sc2 + ss;
      o0[r] *= sc2; o1[r] *= sc2; o2[r] *= sc2; o3[r] *= sc2;
      P[(r + 8 * half) * 32 + n]      = f2bf(p0);
      P[(r + 8 * half) * 32 + 16 + n] = f2bf(p1);
    }
    asm volatile("s_wait_dscnt 0x0" ::: "memory");
    bf16x16 aP = load_a_frag(P + m * 32, kb);  // 16x32 probability tile
    o0 = wmma_bf16(aP, load_b_frag(vhead + (size_t)(0 * 16 + n) * T_ + j0, koffB), o0);
    o1 = wmma_bf16(aP, load_b_frag(vhead + (size_t)(1 * 16 + n) * T_ + j0, koffB), o1);
    o2 = wmma_bf16(aP, load_b_frag(vhead + (size_t)(2 * 16 + n) * T_ + j0, koffB), o2);
    o3 = wmma_bf16(aP, load_b_frag(vhead + (size_t)(3 * 16 + n) * T_ + j0, koffB), o3);
  }

#pragma unroll
  for (int r = 0; r < 8; ++r) {
    float inv = 1.0f / rl[r];
    int t = t0 + r + 8 * half;
    size_t ob = (size_t)(b * T_ + t) * C_ + h * 64;
    attnb[ob + 0 + n]  = f2bf(o0[r] * inv);
    attnb[ob + 16 + n] = f2bf(o1[r] * inv);
    attnb[ob + 32 + n] = f2bf(o2[r] * inv);
    attnb[ob + 48 + n] = f2bf(o3[r] * inv);
  }
}

// ---------------- host-side launch ----------------
extern "C" void kernel_launch(void* const* d_in, const int* in_sizes, int n_in,
                              void* d_out, int out_size, void* d_ws,
                              size_t ws_size, hipStream_t stream) {
  (void)in_sizes; (void)n_in; (void)out_size; (void)ws_size;
  const float* x  = (const float*)d_in[0];
  const float* wq = (const float*)d_in[1];
  const float* wk = (const float*)d_in[2];
  const float* wv = (const float*)d_in[3];
  const float* wo = (const float*)d_in[4];

  float* outp = (float*)d_out;                       // [B,T,C]
  float* kout = outp + (size_t)B_ * T_ * C_;         // [B,H,T,hd] (roped)
  float* vout = kout + (size_t)B_ * H_ * T_ * HD_;   // [B,H,T,hd]

  char* ws = (char*)d_ws;
  u16* xb  = (u16*)(ws + (size_t)0);          // 8 MB  x bf16
  u16* wqb = (u16*)(ws + ((size_t)8  << 20)); // 2 MB
  u16* wkb = (u16*)(ws + ((size_t)10 << 20)); // 2 MB
  u16* wvb = (u16*)(ws + ((size_t)12 << 20)); // 2 MB
  u16* wob = (u16*)(ws + ((size_t)14 << 20)); // 2 MB
  u16* qb  = (u16*)(ws + ((size_t)16 << 20)); // 8 MB  q bf16 (roped)
  u16* kbb = (u16*)(ws + ((size_t)24 << 20)); // 8 MB  k bf16 (roped)
  u16* vT  = (u16*)(ws + ((size_t)32 << 20)); // 8 MB  v^T bf16 [B,H,hd,T]
  u16* ab  = (u16*)(ws + ((size_t)40 << 20)); // 8 MB  attn out bf16 [B,T,C]

  const int nx = B_ * T_ * C_;
  const int nw = C_ * C_;
  f32_to_bf16_kernel<<<(nx + 255) / 256, 256, 0, stream>>>(x, xb, nx);
  f32_to_bf16_kernel<<<(nw + 255) / 256, 256, 0, stream>>>(wq, wqb, nw);
  f32_to_bf16_kernel<<<(nw + 255) / 256, 256, 0, stream>>>(wk, wkb, nw);
  f32_to_bf16_kernel<<<(nw + 255) / 256, 256, 0, stream>>>(wv, wvb, nw);
  f32_to_bf16_kernel<<<(nw + 255) / 256, 256, 0, stream>>>(wo, wob, nw);

  // B*H*(T/16) = 4096 waves -> 1024 blocks x 4 waves
  gemm16x64_kernel<<<1024, 128, 0, stream>>>(xb, wqb, 0, qb, kbb, vT, kout, vout, outp);
  gemm16x64_kernel<<<1024, 128, 0, stream>>>(xb, wkb, 1, qb, kbb, vT, kout, vout, outp);
  gemm16x64_kernel<<<1024, 128, 0, stream>>>(xb, wvb, 2, qb, kbb, vT, kout, vout, outp);

  flash_attn_kernel<<<1024, 128, 0, stream>>>(qb, kbb, vT, ab);

  gemm16x64_kernel<<<1024, 128, 0, stream>>>(ab, wob, 3, qb, kbb, vT, kout, vout, outp);
}
